// ChannelBlock_13417477833042
// MI455X (gfx1250) — matrix operand order repaired and checked by reading, not compiled, not run on hardware
//
#include <hip/hip_runtime.h>
#include <math.h>

#define DIMC     384
#define HEADS    12
#define HEAD_DIM 32
#define HIDDEN   1536
#define BATCH    32
#define HH       28
#define WWID     28
#define NTOK     784            // 28*28
#define MROWS    (BATCH * NTOK) // 25088
#define QKVDIM   (3 * DIMC)     // 1152
#define KSCALE   0.1767766952966369f // 32^-0.5
#define LNEPS    1e-5f

typedef __attribute__((ext_vector_type(16))) __bf16        v16bf;
typedef __attribute__((ext_vector_type(8)))  float         v8f;
typedef __attribute__((ext_vector_type(16))) unsigned short v16s;
typedef __attribute__((ext_vector_type(8)))  unsigned short v8s;

// ---------- helpers ----------
__device__ __forceinline__ unsigned short f2bf(float f) {
  unsigned int u = __builtin_bit_cast(unsigned int, f);
  u += 0x7FFFu + ((u >> 16) & 1u); // round-to-nearest-even
  return (unsigned short)(u >> 16);
}
__device__ __forceinline__ v8f wmma_bf16(v16bf a, v16bf b, v8f c) {
  // D = A(16x32) * B(32x16) + C, f32 accumulate
  return __builtin_amdgcn_wmma_f32_16x16x32_bf16(false, a, false, b, (short)0, c,
                                                 false, false);
}

// ---------- fp32 -> bf16 conversion (weights) ----------
__global__ void cvt_f32_bf16(const float* __restrict__ in,
                             unsigned short* __restrict__ out, int n) {
  int i = blockIdx.x * blockDim.x + threadIdx.x;
  if (i < n) out[i] = f2bf(in[i]);
}

// ---------- depthwise 3x3 conv + residual; output in (B,N,C) fp32 ----------
template <bool IN_BCHW>
__global__ void dwconv_add(const float* __restrict__ x, const float* __restrict__ w,
                           const float* __restrict__ bias, float* __restrict__ out) {
  int tid = blockIdx.x * blockDim.x + threadIdx.x;
  if (tid >= BATCH * NTOK * DIMC) return;
  int c  = tid % DIMC;
  int t  = tid / DIMC;
  int hw = t % NTOK;
  int b  = t / NTOK;
  int h = hw / WWID, wq = hw % WWID;
  float acc = bias[c];
#pragma unroll
  for (int ky = 0; ky < 3; ++ky) {
    int yy = h + ky - 1;
    if (yy < 0 || yy >= HH) continue;
#pragma unroll
    for (int kx = 0; kx < 3; ++kx) {
      int xx = wq + kx - 1;
      if (xx < 0 || xx >= WWID) continue;
      float xv;
      if (IN_BCHW)
        xv = x[(((size_t)b * DIMC + c) * HH + yy) * WWID + xx];
      else
        xv = x[((size_t)b * NTOK + yy * WWID + xx) * DIMC + c];
      acc += xv * w[c * 9 + ky * 3 + kx];
    }
  }
  float center;
  if (IN_BCHW)
    center = x[(((size_t)b * DIMC + c) * HH + h) * WWID + wq];
  else
    center = x[((size_t)b * NTOK + hw) * DIMC + c];
  out[((size_t)b * NTOK + hw) * DIMC + c] = center + acc;
}

// ---------- layernorm over C=384, one wave32 per row, bf16 out ----------
__global__ __launch_bounds__(256) void layernorm_bf16(
    const float* __restrict__ X, const float* __restrict__ g,
    const float* __restrict__ bta, unsigned short* __restrict__ out) {
  int lane = threadIdx.x & 31;
  int wave = threadIdx.x >> 5;
  int row  = blockIdx.x * 8 + wave;
  if (row >= MROWS) return;
  const float* xr = X + (size_t)row * DIMC;
  float v[12];
  float s = 0.f;
#pragma unroll
  for (int j = 0; j < 12; ++j) { v[j] = xr[lane + j * 32]; s += v[j]; }
#pragma unroll
  for (int o = 16; o >= 1; o >>= 1) s += __shfl_xor(s, o, 32);
  float mu = s * (1.0f / DIMC);
  float vs = 0.f;
#pragma unroll
  for (int j = 0; j < 12; ++j) { float d = v[j] - mu; vs += d * d; }
#pragma unroll
  for (int o = 16; o >= 1; o >>= 1) vs += __shfl_xor(vs, o, 32);
  float rs = rsqrtf(vs * (1.0f / DIMC) + LNEPS);
  unsigned short* orow = out + (size_t)row * DIMC;
#pragma unroll
  for (int j = 0; j < 12; ++j) {
    int c   = lane + j * 32;
    orow[c] = f2bf((v[j] - mu) * rs * g[c] + bta[c]);
  }
}

// ---------- generic bf16 GEMM: out(M,N) = A(M,K) * W(N,K)^T  (+ epilogue) ----
// wave-level: one wave computes a 32(M) x 64(N) tile; block = 8 waves = 256 M.
enum { EPI_QKV = 0, EPI_BIAS_RES = 1, EPI_BIAS_GELU = 2, EPI_BIAS_RES_TRANS = 3 };

template <int EPI>
__global__ __launch_bounds__(256) void gemm_bf16(
    const unsigned short* __restrict__ A, const unsigned short* __restrict__ W,
    const float* __restrict__ bias, const float* __restrict__ res,
    float* __restrict__ outF, unsigned short* __restrict__ outB, int Nout, int K) {
  int lane = threadIdx.x & 31;
  int wave = threadIdx.x >> 5;
  int m0   = (blockIdx.y * 8 + wave) * 32;
  int n0   = blockIdx.x * 64;
  int hi   = lane >> 4;   // which half of the wave
  int lcol = lane & 15;

  v8f acc[2][4];
#pragma unroll
  for (int mt = 0; mt < 2; ++mt)
#pragma unroll
    for (int t = 0; t < 4; ++t) acc[mt][t] = v8f{};

  const unsigned short* arow0 = A + (size_t)(m0 + lcol) * K;
  const unsigned short* arow1 = A + (size_t)(m0 + 16 + lcol) * K;
  for (int k0 = 0; k0 < K; k0 += 32) {
    // A fragments (16x32): lane holds row (lcol),
    // elems 0..15 -> K = k0 + hi*16 + j (contiguous, matches B)
    v16bf af[2];
#pragma unroll
    for (int mt = 0; mt < 2; ++mt) {
      const unsigned short* ar = mt ? arow1 : arow0;
      v8s a0 = *(const v8s*)(ar + k0 + hi * 16);
      v8s a1 = *(const v8s*)(ar + k0 + hi * 16 + 8);
      v16s at;
#pragma unroll
      for (int j = 0; j < 8; ++j) { at[j] = a0[j]; at[8 + j] = a1[j]; }
      af[mt] = __builtin_bit_cast(v16bf, at);
    }
#pragma unroll
    for (int t = 0; t < 4; ++t) {
      // B fragment (32x16, K-major per lane): lane = column n0+t*16+lcol,
      // 16 contiguous K values starting at k0 + hi*16 -> one 32B load
      v16bf bf = __builtin_bit_cast(
          v16bf, *(const v16s*)(W + (size_t)(n0 + t * 16 + lcol) * K + k0 + hi * 16));
      acc[0][t] = wmma_bf16(af[0], bf, acc[0][t]);
      acc[1][t] = wmma_bf16(af[1], bf, acc[1][t]);
    }
  }

  // epilogue: C/D layout -> VGPR i: lanes0-15 M=i, lanes16-31 M=i+8
#pragma unroll
  for (int mt = 0; mt < 2; ++mt) {
#pragma unroll
    for (int t = 0; t < 4; ++t) {
      int col = n0 + t * 16 + lcol;
#pragma unroll
      for (int i = 0; i < 8; ++i) {
        int   m   = m0 + mt * 16 + i + hi * 8;
        float val = acc[mt][t][i];
        if (EPI == EPI_QKV) {
          if (col >= DIMC && col < 2 * DIMC) val *= KSCALE; // scale k-columns
          outB[(size_t)m * Nout + col] = f2bf(val);
        } else if (EPI == EPI_BIAS_RES) {
          val += bias[col] + res[(size_t)m * Nout + col];
          outF[(size_t)m * Nout + col] = val;
        } else if (EPI == EPI_BIAS_GELU) {
          val += bias[col];
          val = 0.5f * val * (1.0f + erff(val * 0.70710678118654752f));
          outB[(size_t)m * Nout + col] = f2bf(val);
        } else { // EPI_BIAS_RES_TRANS: residual + write fp32 in (B,C,H,W)
          val += bias[col] + res[(size_t)m * Nout + col];
          int b = m / NTOK, n = m % NTOK;
          outF[((size_t)b * DIMC + col) * NTOK + n] = val;
        }
      }
    }
  }
}

// ---------- per-(b,h): S = sum_n k[n,:]^T v[n,:]  (32x32), softmax rows ------
__global__ __launch_bounds__(32) void xca_attn_softmax(
    const unsigned short* __restrict__ qkv, unsigned short* __restrict__ attn) {
  int bh   = blockIdx.x; // b*HEADS + h
  int b    = bh / HEADS;
  int h    = bh % HEADS;
  int lane = threadIdx.x & 31;
  int hi   = lane >> 4;
  int lc   = lane & 15;
  __shared__ float S[32][33];

  const size_t baseK = (size_t)b * NTOK * QKVDIM + DIMC + h * HEAD_DIM;
  const size_t baseV = (size_t)b * NTOK * QKVDIM + 2 * DIMC + h * HEAD_DIM;

  v8f acc[2][2];
#pragma unroll
  for (int dt = 0; dt < 2; ++dt)
#pragma unroll
    for (int et = 0; et < 2; ++et) acc[dt][et] = v8f{};

  // ---- 24 full, unguarded 32-row chunks ----
  for (int n0 = 0; n0 < 768; n0 += 32) {
    v16bf afr[2];
#pragma unroll
    for (int dt = 0; dt < 2; ++dt) { // A = K^T : A[d][n] = k[n][d]
      v16s t;
#pragma unroll
      for (int j = 0; j < 16; ++j) {
        int kk = (j < 8) ? (hi * 8 + j) : (16 + hi * 8 + (j - 8));
        t[j]   = qkv[baseK + (size_t)(n0 + kk) * QKVDIM + dt * 16 + lc];
      }
      afr[dt] = __builtin_bit_cast(v16bf, t);
    }
#pragma unroll
    for (int et = 0; et < 2; ++et) { // B = V : B[n][e]
      v16s t;
#pragma unroll
      for (int j = 0; j < 16; ++j)
        t[j] = qkv[baseV + (size_t)(n0 + ((j < 8) ? (hi * 8 + j) : (16 + hi * 8 + (j - 8)))) * QKVDIM + et * 16 + lc];
      v16bf bfr  = __builtin_bit_cast(v16bf, t);
      acc[0][et] = wmma_bf16(afr[0], bfr, acc[0][et]);
      acc[1][et] = wmma_bf16(afr[1], bfr, acc[1][et]);
    }
  }

  // ---- tail chunk n0=768: rows 768..783 valid (16 rows) ----
  {
    const int n0 = 768;
    v16bf afr[2];
#pragma unroll
    for (int dt = 0; dt < 2; ++dt) {
      // valid elements are exactly j<8 (kk = hi*8+j < 16) for both halves;
      // j>=8 (kk >= 16) is past the end -> compile-time zeros, no branches.
      v16s t;
#pragma unroll
      for (int j = 0; j < 8; ++j)
        t[j] = qkv[baseK + (size_t)(n0 + hi * 8 + j) * QKVDIM + dt * 16 + lc];
#pragma unroll
      for (int j = 8; j < 16; ++j) t[j] = 0;
      afr[dt] = __builtin_bit_cast(v16bf, t);
    }
#pragma unroll
    for (int et = 0; et < 2; ++et) {
      // elements j<8 valid for both halves; j>=8 (kk >= 16) reads stay inside
      // our workspace, then get zero-selected.
      v16s t;
#pragma unroll
      for (int j = 0; j < 16; ++j) {
        unsigned short r =
            qkv[baseV + (size_t)(n0 + ((j < 8) ? (hi * 8 + j) : (16 + hi * 8 + (j - 8)))) * QKVDIM + et * 16 + lc];
        t[j] = (j >= 8) ? (unsigned short)0 : r;
      }
      v16bf bfr  = __builtin_bit_cast(v16bf, t);
      acc[0][et] = wmma_bf16(afr[0], bfr, acc[0][et]);
      acc[1][et] = wmma_bf16(afr[1], bfr, acc[1][et]);
    }
  }

  // dump 32x32 to LDS
#pragma unroll
  for (int dt = 0; dt < 2; ++dt)
#pragma unroll
    for (int et = 0; et < 2; ++et)
#pragma unroll
      for (int i = 0; i < 8; ++i)
        S[dt * 16 + i + hi * 8][et * 16 + lc] = acc[dt][et][i];
  __syncthreads();

  // softmax over e for row = lane
  float mx = -1e30f;
#pragma unroll
  for (int e = 0; e < 32; ++e) mx = fmaxf(mx, S[lane][e]);
  float ex[32];
  float sum = 0.f;
#pragma unroll
  for (int e = 0; e < 32; ++e) { ex[e] = expf(S[lane][e] - mx); sum += ex[e]; }
  float inv = 1.0f / sum;
  unsigned short* arow = attn + ((size_t)bh * 32 + lane) * 32;
#pragma unroll
  for (int e = 0; e < 32; ++e) arow[e] = f2bf(ex[e] * inv);
}

// ---------- out[n, h*32+d] = sum_e attn[d,e] * q[n,e]  (12 waves = 12 heads) --
__global__ __launch_bounds__(384) void xca_attn_out(
    const unsigned short* __restrict__ qkv, const unsigned short* __restrict__ attn,
    unsigned short* __restrict__ aout) {
  int h    = threadIdx.x >> 5; // one wave per head
  int lane = threadIdx.x & 31;
  int hi   = lane >> 4;
  int lc   = lane & 15;
  int b    = blockIdx.y;
  int n0   = blockIdx.x * 16;
  int bh   = b * HEADS + h;

  const size_t baseQ = (size_t)b * NTOK * QKVDIM + h * HEAD_DIM;
  // A = Q tile (16 tokens x 32 e)
  v16s t;
  int  n = n0 + lc;
#pragma unroll
  for (int j = 0; j < 16; ++j) {
    int e = hi * 16 + j;
    t[j]  = qkv[baseQ + (size_t)n * QKVDIM + e];
  }
  v16bf af = __builtin_bit_cast(v16bf, t);

  v8f acc[2];
#pragma unroll
  for (int dt = 0; dt < 2; ++dt) acc[dt] = v8f{};
#pragma unroll
  for (int dt = 0; dt < 2; ++dt) {
    // B[e][d] = attn[dt*16+lc][e]; 16 contiguous bf16 -> 32B vector load
    v16bf bfr = __builtin_bit_cast(
        v16bf, *(const v16s*)(attn + ((size_t)bh * 32 + dt * 16 + lc) * 32 + hi * 16));
    acc[dt] = wmma_bf16(af, bfr, acc[dt]);
  }
#pragma unroll
  for (int dt = 0; dt < 2; ++dt)
#pragma unroll
    for (int i = 0; i < 8; ++i) {
      int nn = n0 + i + hi * 8;
      aout[((size_t)b * NTOK + nn) * DIMC + h * HEAD_DIM + dt * 16 + lc] =
          f2bf(acc[dt][i]);
    }
}

// ---------------------------------------------------------------------------
extern "C" void kernel_launch(void* const* d_in, const int* in_sizes, int n_in,
                              void* d_out, int out_size, void* d_ws, size_t ws_size,
                              hipStream_t stream) {
  const float* x      = (const float*)d_in[0];
  const float* cpe1_w = (const float*)d_in[1];
  const float* cpe1_b = (const float*)d_in[2];
  const float* n1g    = (const float*)d_in[3];
  const float* n1b    = (const float*)d_in[4];
  const float* qkv_w  = (const float*)d_in[5];
  const float* proj_w = (const float*)d_in[6];
  const float* proj_b = (const float*)d_in[7];
  const float* cpe2_w = (const float*)d_in[8];
  const float* cpe2_b = (const float*)d_in[9];
  const float* n2g    = (const float*)d_in[10];
  const float* n2b    = (const float*)d_in[11];
  const float* fc1_w  = (const float*)d_in[12];
  const float* fc1_b  = (const float*)d_in[13];
  const float* fc2_w  = (const float*)d_in[14];
  const float* fc2_b  = (const float*)d_in[15];
  float*       out    = (float*)d_out;

  char*  ws  = (char*)d_ws;
  size_t off = 0;
  auto   alloc = [&](size_t bytes) -> void* {
    void* p = ws + off;
    off     = (off + bytes + 255) & ~(size_t)255;
    return p;
  };

  unsigned short* wqkv  = (unsigned short*)alloc((size_t)QKVDIM * DIMC * 2);
  unsigned short* wproj = (unsigned short*)alloc((size_t)DIMC * DIMC * 2);
  unsigned short* wfc1  = (unsigned short*)alloc((size_t)HIDDEN * DIMC * 2);
  unsigned short* wfc2  = (unsigned short*)alloc((size_t)DIMC * HIDDEN * 2);
  float*          xA    = (float*)alloc((size_t)MROWS * DIMC * 4); // x1 / x2 (in place)
  float*          xB    = (float*)alloc((size_t)MROWS * DIMC * 4); // x3
  unsigned short* cur   = (unsigned short*)alloc((size_t)MROWS * DIMC * 2);
  unsigned short* qkvb  = (unsigned short*)alloc((size_t)MROWS * HIDDEN * 2); // qkv, reused as y1
  unsigned short* attnb = (unsigned short*)alloc((size_t)BATCH * HEADS * 32 * 32 * 2);
  unsigned short* aoutb = (unsigned short*)alloc((size_t)MROWS * DIMC * 2);
  unsigned short* y1    = qkvb;

  // 1. weights -> bf16
  cvt_f32_bf16<<<(QKVDIM * DIMC + 255) / 256, 256, 0, stream>>>(qkv_w, wqkv,
                                                                QKVDIM * DIMC);
  cvt_f32_bf16<<<(DIMC * DIMC + 255) / 256, 256, 0, stream>>>(proj_w, wproj,
                                                              DIMC * DIMC);
  cvt_f32_bf16<<<(HIDDEN * DIMC + 255) / 256, 256, 0, stream>>>(fc1_w, wfc1,
                                                                HIDDEN * DIMC);
  cvt_f32_bf16<<<(DIMC * HIDDEN + 255) / 256, 256, 0, stream>>>(fc2_w, wfc2,
                                                                DIMC * HIDDEN);

  // 2. x1 = x + dwconv1(x), BCHW -> (B,N,C) fp32
  dwconv_add<true><<<(MROWS * DIMC + 255) / 256, 256, 0, stream>>>(x, cpe1_w,
                                                                   cpe1_b, xA);
  // 3. cur = LN(x1) -> bf16
  layernorm_bf16<<<MROWS / 8, 256, 0, stream>>>(xA, n1g, n1b, cur);
  // 4. qkv = cur @ qkv_w^T (k scaled), bf16
  gemm_bf16<EPI_QKV><<<dim3(QKVDIM / 64, MROWS / 256), 256, 0, stream>>>(
      cur, wqkv, nullptr, nullptr, nullptr, qkvb, QKVDIM, DIMC);
  // 5. attn = softmax(K^T V) per (b,h)
  xca_attn_softmax<<<BATCH * HEADS, 32, 0, stream>>>(qkvb, attnb);
  // 6. aout = Q @ attn^T per head
  xca_attn_out<<<dim3(NTOK / 16, BATCH), 384, 0, stream>>>(qkvb, attnb, aoutb);
  // 7. x2 = x1 + aout @ proj_w^T + proj_b (in place in xA)
  gemm_bf16<EPI_BIAS_RES><<<dim3(DIMC / 64, MROWS / 256), 256, 0, stream>>>(
      aoutb, wproj, proj_b, xA, xA, nullptr, DIMC, DIMC);
  // 8. x3 = x2 + dwconv2(x2), BNC -> BNC
  dwconv_add<false><<<(MROWS * DIMC + 255) / 256, 256, 0, stream>>>(xA, cpe2_w,
                                                                    cpe2_b, xB);
  // 9. cur = LN(x3) -> bf16
  layernorm_bf16<<<MROWS / 8, 256, 0, stream>>>(xB, n2g, n2b, cur);
  // 10. y1 = gelu(cur @ fc1_w^T + fc1_b) -> bf16
  gemm_bf16<EPI_BIAS_GELU><<<dim3(HIDDEN / 64, MROWS / 256), 256, 0, stream>>>(
      cur, wfc1, fc1_b, nullptr, nullptr, y1, HIDDEN, DIMC);
  // 11. out(BCHW) = transpose(x3 + y1 @ fc2_w^T + fc2_b)
  gemm_bf16<EPI_BIAS_RES_TRANS><<<dim3(DIMC / 64, MROWS / 256), 256, 0, stream>>>(
      y1, wfc2, fc2_b, xB, out, nullptr, DIMC, HIDDEN);
}
